// DropEdgeGAT_15522011808338
// MI455X (gfx1250) — compile-verified, matrix operand order checked by reading, not run on hardware
//
#include <hip/hip_runtime.h>
#include <hip/hip_bf16.h>

typedef unsigned short u16;
typedef unsigned int   u32;

typedef __attribute__((ext_vector_type(16))) __bf16 v16bf;
typedef __attribute__((ext_vector_type(8)))  float  v8f;

#define N_NODES 50000
#define N_EDGES 800000
#define F_IN    128
#define HID     64
#define HEADS   4
#define DMODEL  256   // HEADS*HID
#define NCLS    10
#define NCLS_P  16    // padded
#define LRELU   0.2f

// ---------------------------------------------------------------------------
// helpers
// ---------------------------------------------------------------------------
__device__ __forceinline__ u16 f2bf(float f) {
    u32 u = __float_as_uint(f);
    u32 r = (u + 0x7fffu + ((u >> 16) & 1u)) >> 16;   // round-to-nearest-even
    return (u16)r;
}

__device__ __forceinline__ float atomicMaxF32(float* addr, float val) {
    int* ai = (int*)addr;
    int old = __float_as_int(*addr);
    while (__int_as_float(old) < val) {
        int assumed = old;
        old = atomicCAS(ai, assumed, __float_as_int(val));
        if (old == assumed) break;
    }
    return __int_as_float(old);
}

// ---------------------------------------------------------------------------
// Tensor Data Mover path (guarded: falls back to vector loads if unavailable)
// This toolchain exposes the 6-arg builtin:
//   (uint32x4 g0, int32x8 g1, int32x4 g2, int32x4 g3, int32x8, int cpol)
// ---------------------------------------------------------------------------
#if __has_builtin(__builtin_amdgcn_tensor_load_to_lds) && __has_builtin(__builtin_amdgcn_s_wait_tensorcnt)
#define HAVE_TDM 1
typedef unsigned int u32x4 __attribute__((ext_vector_type(4)));
typedef int          i32x8 __attribute__((ext_vector_type(8)));
typedef int          i32x4 __attribute__((ext_vector_type(4)));

// 2D bf16 tile DMA: global (row stride = stride0 elements) -> LDS, packed
// row-major tile (tile_d0 contiguous elements per row). D# per ISA 8.3/8.4.
__device__ __forceinline__ void tdm_load_2d(u32 lds_off, const u16* gptr,
                                            u32 tensor_d0, u32 tensor_d1,
                                            u32 tile_d0, u32 tile_d1, u32 stride0) {
    unsigned long long ga = (unsigned long long)(uintptr_t)gptr;
    u32x4 g0;
    g0[0] = 1u;                                            // count=1 user D#
    g0[1] = lds_off;                                       // lds_addr (bytes)
    g0[2] = (u32)ga;                                       // global_addr lo
    g0[3] = ((u32)(ga >> 32) & 0x01FFFFFFu) | 0x80000000u; // global_addr hi | type=2
    i32x8 g1;
    g1[0] = (int)(1u << 16);                               // data_size=1 (2 bytes)
    g1[1] = (int)((tensor_d0 & 0xFFFFu) << 16);            // tensor_dim0[15:0]
    g1[2] = (int)((tensor_d0 >> 16) | ((tensor_d1 & 0xFFFFu) << 16));
    g1[3] = (int)((tensor_d1 >> 16) | (tile_d0 << 16));    // tile_dim0
    g1[4] = (int)(tile_d1 & 0xFFFFu);                      // tile_dim1 (tile_dim2=0)
    g1[5] = (int)stride0;                                  // tensor_dim0_stride lo
    g1[6] = 0;
    g1[7] = 0;
    i32x4 z4 = {0, 0, 0, 0};
    i32x8 z8 = {0, 0, 0, 0, 0, 0, 0, 0};
    __builtin_amdgcn_tensor_load_to_lds(g0, g1, z4, z4, z8, 0);
}
#else
#define HAVE_TDM 0
#endif

// ---------------------------------------------------------------------------
// conversion / setup kernels
// ---------------------------------------------------------------------------
__global__ void k_f32_to_bf16(const float* __restrict__ in, u16* __restrict__ out, int n) {
    for (int i = blockIdx.x * blockDim.x + threadIdx.x; i < n; i += gridDim.x * blockDim.x)
        out[i] = f2bf(in[i]);
}

// W[K][NcSrc] f32 -> Bt[NcDst][K] bf16 (transposed, zero-padded cols)
__global__ void k_transpose_bf16(const float* __restrict__ W, u16* __restrict__ Bt,
                                 int K, int NcSrc, int NcDst) {
    int total = NcDst * K;
    for (int i = blockIdx.x * blockDim.x + threadIdx.x; i < total; i += gridDim.x * blockDim.x) {
        int n = i / K, k = i % K;
        float v = (n < NcSrc) ? W[(size_t)k * NcSrc + n] : 0.f;
        Bt[i] = f2bf(v);
    }
}

__global__ void k_init(float* __restrict__ p, float v, int n) {
    for (int i = blockIdx.x * blockDim.x + threadIdx.x; i < n; i += gridDim.x * blockDim.x)
        p[i] = v;
}

// fold  M[k,h] = sum_c We[k, h*64+c] * atte[h,c]   (4x4 per layer; 32 threads)
__global__ void k_fold_M(const float* __restrict__ We1, const float* __restrict__ ae1,
                         const float* __restrict__ We2, const float* __restrict__ ae2,
                         float* __restrict__ M) {
    int t = threadIdx.x;
    if (t >= 32) return;
    const float* We = (t < 16) ? We1 : We2;
    const float* ae = (t < 16) ? ae1 : ae2;
    int idx = t & 15, k = idx >> 2, h = idx & 3;
    float s = 0.f;
    for (int c = 0; c < HID; ++c)
        s += We[k * DMODEL + h * HID + c] * ae[h * HID + c];
    M[t] = s;
}

// ---------------------------------------------------------------------------
// edge features + 2-layer MLP encoder  (one wave per edge)
// ---------------------------------------------------------------------------
__global__ __launch_bounds__(256)
void k_edge_encode(const float* __restrict__ x, const int* __restrict__ src,
                   const int* __restrict__ dst,
                   const float* __restrict__ Wee1, const float* __restrict__ bee1,
                   const float* __restrict__ Wee2, const float* __restrict__ bee2,
                   float* __restrict__ ea, int E) {
    int wid  = (blockIdx.x * blockDim.x + threadIdx.x) >> 5;
    int lane = threadIdx.x & 31;
    if (wid >= E) return;
    int s = src[wid], d = dst[wid];
    float4 a = ((const float4*)(x + (size_t)s * F_IN))[lane];
    float4 b = ((const float4*)(x + (size_t)d * F_IN))[lane];
    float acc = a.x * b.x + a.y * b.y + a.z * b.z + a.w * b.w;
    #pragma unroll
    for (int off = 16; off; off >>= 1) acc += __shfl_down(acc, off, 32);
    acc = __shfl(acc, 0, 32);
    float sv = acc * (1.f / (float)F_IN);
    float p0 = sv, p1 = sv * sv, p2 = p1 * sv, p3 = p2 * sv;
    float h0 = bee1[lane], h1 = bee1[lane + 32];
    h0 += p0 * Wee1[0 * HID + lane] + p1 * Wee1[1 * HID + lane]
        + p2 * Wee1[2 * HID + lane] + p3 * Wee1[3 * HID + lane];
    h1 += p0 * Wee1[0 * HID + lane + 32] + p1 * Wee1[1 * HID + lane + 32]
        + p2 * Wee1[2 * HID + lane + 32] + p3 * Wee1[3 * HID + lane + 32];
    h0 = fmaxf(h0, 0.f); h1 = fmaxf(h1, 0.f);
    #pragma unroll
    for (int h = 0; h < HEADS; ++h) {
        float ph = h0 * Wee2[lane * HEADS + h] + h1 * Wee2[(lane + 32) * HEADS + h];
        #pragma unroll
        for (int off = 16; off; off >>= 1) ph += __shfl_down(ph, off, 32);
        if (lane == 0) ea[(size_t)wid * HEADS + h] = ph + bee2[h];
    }
}

// ---------------------------------------------------------------------------
// bf16 WMMA GEMM:  C[M,Nc] (f32) = A[M,K] (bf16 row-major) @ B (given as
// Bt[Nc][K] bf16, k-contiguous).  Block tile 128x64, 8 waves x (16 rows, 4
// col-tiles), K chunks of 32.  B tile staged ONCE (whole K fits in LDS);
// A chunks staged per iteration.  TDM DMA when available (OOB rows/cols
// zero-filled by hardware), else clamped vector loads.
// ---------------------------------------------------------------------------
#define TM 128
#define TN 64
#define TK 32
#define KMAX 256

__global__ __launch_bounds__(256)
void k_wmma_gemm(const u16* __restrict__ A, const u16* __restrict__ Bt,
                 float* __restrict__ C, int M, int Nc, int K) {
    __shared__ __align__(16) u16 lA[TM * TK];      // 8 KB
    __shared__ __align__(16) u16 lB[TN * KMAX];    // 32 KB (rows of length K used)

    int wave = threadIdx.x >> 5;
    int lane = threadIdx.x & 31;
    int bm = blockIdx.x * TM;
    int bn = blockIdx.y * TN;

#if HAVE_TDM
    if (threadIdx.x < 32) {   // wave 0 only (uniform branch; single DMA issue)
        tdm_load_2d((u32)(uintptr_t)lB, Bt + (size_t)bn * K,
                    (u32)K, (u32)(Nc - bn), (u32)K, (u32)TN, (u32)K);
    }
#else
    for (int i = threadIdx.x; i < TN * (K / 8); i += 256) {
        int n  = i / (K / 8);
        int kc = (i % (K / 8)) * 8;
        int gn = min(bn + n, Nc - 1);               // clamp: no divergence
        *(uint4*)&lB[n * K + kc] = *(const uint4*)(Bt + (size_t)gn * K + kc);
    }
#endif

    v8f acc[4] = {};

    for (int k0 = 0; k0 < K; k0 += TK) {
#if HAVE_TDM
        if (threadIdx.x < 32) {
            tdm_load_2d((u32)(uintptr_t)lA, A + (size_t)bm * K + k0,
                        (u32)(K - k0), (u32)(M - bm), (u32)TK, (u32)TM, (u32)K);
            __builtin_amdgcn_s_wait_tensorcnt(0);   // covers B DMA too (in-order)
        }
#else
        for (int i = threadIdx.x; i < TM * (TK / 2); i += 256) {
            int r  = i >> 4;                        // TK/2 = 16 uints per row
            int kk = (i & 15) * 2;
            int gr = min(bm + r, M - 1);            // clamp: no divergence
            *(u32*)&lA[r * TK + kk] = *(const u32*)(A + (size_t)gr * K + k0 + kk);
        }
#endif
        __syncthreads();

        // A fragment (ISA 7.12.2, 16-bit A 16x32): lane<16 holds row=lane,
        // K in {0..7,16..23}; lane>=16 holds row=lane-16, K in {8..15,24..31}
        union { v16bf v; u32 u[8]; } fa;
        {
            const u16* base = &lA[(wave * 16 + (lane & 15)) * TK];
            int kb = (lane >> 4) * 8;
            #pragma unroll
            for (int j = 0; j < 4; ++j) fa.u[j]     = *(const u32*)(base + kb + 2 * j);
            #pragma unroll
            for (int j = 0; j < 4; ++j) fa.u[4 + j] = *(const u32*)(base + kb + 16 + 2 * j);
        }
        #pragma unroll
        for (int t = 0; t < 4; ++t) {
            union { v16bf v; u32 u[8]; } fb;
            const u16* base = &lB[(t * 16 + (lane & 15)) * K + k0];
            int kb = (lane >> 4) * 8;
            #pragma unroll
            for (int j = 0; j < 4; ++j) fb.u[j]     = *(const u32*)(base + kb + 2 * j);
            #pragma unroll
            for (int j = 0; j < 4; ++j) fb.u[4 + j] = *(const u32*)(base + kb + 16 + 2 * j);
            acc[t] = __builtin_amdgcn_wmma_f32_16x16x32_bf16(
                         false, fa.v, false, fb.v, (short)0, acc[t], false, false);
        }
        __syncthreads();
    }

    // C/D layout: lane holds col = lane&15 (+16*t); vgpr v -> row v + 8*(lane>=16)
    int rowBase = bm + wave * 16;
    int colIn = lane & 15;
    int rofs  = (lane >> 4) * 8;
    #pragma unroll
    for (int t = 0; t < 4; ++t) {
        int gc = bn + t * 16 + colIn;
        if (gc >= Nc) continue;
        #pragma unroll
        for (int v = 0; v < 8; ++v) {
            int gr = rowBase + rofs + v;
            if (gr < M) C[(size_t)gr * Nc + gc] = acc[t][v];
        }
    }
}

// ---------------------------------------------------------------------------
// per-node attention dots:  asrc[n,h] = sum_c XL[n,h*64+c]*a_s[h,c]  (wave/node)
// ---------------------------------------------------------------------------
__global__ __launch_bounds__(256)
void k_node_alpha(const float* __restrict__ XL, const float* __restrict__ a_s,
                  const float* __restrict__ a_d, float* __restrict__ asrc,
                  float* __restrict__ adst, int N) {
    int wid  = (blockIdx.x * blockDim.x + threadIdx.x) >> 5;
    int lane = threadIdx.x & 31;
    if (wid >= N) return;
    const float* row = XL + (size_t)wid * DMODEL;
    float accS[4] = {0.f, 0.f, 0.f, 0.f};
    float accD[4] = {0.f, 0.f, 0.f, 0.f};
    #pragma unroll
    for (int i = 0; i < 8; ++i) {
        int idx = i * 32 + lane;
        float v  = row[idx];
        float vs = v * a_s[idx];
        float vd = v * a_d[idx];
        int h = idx >> 6;
        #pragma unroll
        for (int hh = 0; hh < 4; ++hh) {
            accS[hh] += (hh == h) ? vs : 0.f;
            accD[hh] += (hh == h) ? vd : 0.f;
        }
    }
    #pragma unroll
    for (int h = 0; h < 4; ++h) {
        float s = accS[h], d = accD[h];
        #pragma unroll
        for (int off = 16; off; off >>= 1) {
            s += __shfl_down(s, off, 32);
            d += __shfl_down(d, off, 32);
        }
        if (lane == 0) {
            asrc[(size_t)wid * 4 + h] = s;
            adst[(size_t)wid * 4 + h] = d;
        }
    }
}

// ---------------------------------------------------------------------------
// per-edge alpha = lrelu(asrc[src]+adst[dst]+ea@M) ; segment max via atomics
// ---------------------------------------------------------------------------
__global__ void k_edge_alpha(const int* __restrict__ src, const int* __restrict__ dst,
                             const float* __restrict__ asrc, const float* __restrict__ adst,
                             const float* __restrict__ ea, const float* __restrict__ M,
                             float* __restrict__ alpha, float* __restrict__ amax, int E) {
    int e = blockIdx.x * blockDim.x + threadIdx.x;
    if (e >= E) return;
    int s = src[e], d = dst[e];
    float e0 = ea[(size_t)e * 4 + 0], e1 = ea[(size_t)e * 4 + 1];
    float e2 = ea[(size_t)e * 4 + 2], e3 = ea[(size_t)e * 4 + 3];
    #pragma unroll
    for (int h = 0; h < 4; ++h) {
        float a = asrc[(size_t)s * 4 + h] + adst[(size_t)d * 4 + h]
                + e0 * M[0 * 4 + h] + e1 * M[1 * 4 + h]
                + e2 * M[2 * 4 + h] + e3 * M[3 * 4 + h];
        a = (a > 0.f) ? a : LRELU * a;
        alpha[(size_t)e * 4 + h] = a;
        atomicMaxF32(&amax[(size_t)d * 4 + h], a);
    }
}

// exp(alpha - amax[dst]) in place; denom accumulate
__global__ void k_edge_exp(const int* __restrict__ dst, float* __restrict__ alpha,
                           const float* __restrict__ amax, float* __restrict__ denom, int E) {
    int e = blockIdx.x * blockDim.x + threadIdx.x;
    if (e >= E) return;
    int d = dst[e];
    #pragma unroll
    for (int h = 0; h < 4; ++h) {
        float m = amax[(size_t)d * 4 + h];
        m = (m < -1e30f) ? 0.f : m;
        float v = __expf(alpha[(size_t)e * 4 + h] - m);
        alpha[(size_t)e * 4 + h] = v;
        atomicAdd(&denom[(size_t)d * 4 + h], v);
    }
}

// weighted scatter-add aggregation (wave per edge)
__global__ __launch_bounds__(256)
void k_edge_agg(const int* __restrict__ src, const int* __restrict__ dst,
                const float* __restrict__ alpha, const float* __restrict__ denom,
                const float* __restrict__ XL, float* __restrict__ AGG, int E) {
    int wid  = (blockIdx.x * blockDim.x + threadIdx.x) >> 5;
    int lane = threadIdx.x & 31;
    if (wid >= E) return;
    int s = src[wid], d = dst[wid];
    const float* xr = XL + (size_t)s * DMODEL;
    float* orow = AGG + (size_t)d * DMODEL;
    __builtin_prefetch(xr + lane * 8, 0, 1);        // global_prefetch: overlap gather
    float c0 = alpha[(size_t)wid * 4 + 0] / fmaxf(denom[(size_t)d * 4 + 0], 1e-16f);
    float c1 = alpha[(size_t)wid * 4 + 1] / fmaxf(denom[(size_t)d * 4 + 1], 1e-16f);
    float c2 = alpha[(size_t)wid * 4 + 2] / fmaxf(denom[(size_t)d * 4 + 2], 1e-16f);
    float c3 = alpha[(size_t)wid * 4 + 3] / fmaxf(denom[(size_t)d * 4 + 3], 1e-16f);
    #pragma unroll
    for (int i = 0; i < 8; ++i) {
        int idx = i * 32 + lane;
        int h = idx >> 6;
        float coef = (h == 0) ? c0 : (h == 1) ? c1 : (h == 2) ? c2 : c3;
        atomicAdd(&orow[idx], xr[idx] * coef);
    }
}

// out = elu(AGG + b), stored as bf16 for the next GEMM
__global__ void k_bias_elu_bf16(const float* __restrict__ AGG, const float* __restrict__ b,
                                u16* __restrict__ hbf, int total) {
    for (int i = blockIdx.x * blockDim.x + threadIdx.x; i < total; i += gridDim.x * blockDim.x) {
        int c = i & (DMODEL - 1);
        float v = AGG[i] + b[c];
        v = (v > 0.f) ? v : (__expf(v) - 1.f);
        hbf[i] = f2bf(v);
    }
}

// log-softmax over 10 classes
__global__ void k_logsoftmax(const float* __restrict__ logits, const float* __restrict__ bc,
                             float* __restrict__ out, int N) {
    int n = blockIdx.x * blockDim.x + threadIdx.x;
    if (n >= N) return;
    float l[NCLS];
    float mx = -3.4e38f;
    #pragma unroll
    for (int j = 0; j < NCLS; ++j) {
        l[j] = logits[(size_t)n * NCLS_P + j] + bc[j];
        mx = fmaxf(mx, l[j]);
    }
    float s = 0.f;
    #pragma unroll
    for (int j = 0; j < NCLS; ++j) s += __expf(l[j] - mx);
    float ls = mx + __logf(s);
    #pragma unroll
    for (int j = 0; j < NCLS; ++j) out[(size_t)n * NCLS + j] = l[j] - ls;
}

// ---------------------------------------------------------------------------
// launcher
// ---------------------------------------------------------------------------
extern "C" void kernel_launch(void* const* d_in, const int* in_sizes, int n_in,
                              void* d_out, int out_size, void* d_ws, size_t ws_size,
                              hipStream_t stream) {
    const float* x      = (const float*)d_in[0];
    const int*   eidx   = (const int*)d_in[1];
    const float* Wee1   = (const float*)d_in[2];
    const float* bee1   = (const float*)d_in[3];
    const float* Wee2   = (const float*)d_in[4];
    const float* bee2   = (const float*)d_in[5];
    const float* W1     = (const float*)d_in[6];
    const float* as1    = (const float*)d_in[7];
    const float* ad1    = (const float*)d_in[8];
    const float* We1    = (const float*)d_in[9];
    const float* ae1    = (const float*)d_in[10];
    const float* b1     = (const float*)d_in[11];
    const float* W2     = (const float*)d_in[12];
    const float* as2    = (const float*)d_in[13];
    const float* ad2    = (const float*)d_in[14];
    const float* We2    = (const float*)d_in[15];
    const float* ae2    = (const float*)d_in[16];
    const float* b2     = (const float*)d_in[17];
    const float* Wc     = (const float*)d_in[18];
    const float* bc     = (const float*)d_in[19];
    float* out = (float*)d_out;

    const int* src = eidx;
    const int* dst = eidx + N_EDGES;

    // ---- workspace carve-up -------------------------------------------------
    char* ws = (char*)d_ws;
    size_t o = 0;
    auto carve = [&](size_t bytes) -> char* {
        char* p = ws + o;
        o = (o + bytes + 255) & ~(size_t)255;
        return p;
    };
    u16*   xbf   = (u16*)  carve((size_t)N_NODES * F_IN * 2);
    u16*   W1t   = (u16*)  carve((size_t)DMODEL * F_IN * 2);      // [256][128]
    u16*   W2t   = (u16*)  carve((size_t)DMODEL * DMODEL * 2);    // [256][256]
    u16*   Wct   = (u16*)  carve((size_t)NCLS_P * DMODEL * 2);    // [16][256]
    float* Mf    = (float*)carve(32 * 4);
    float* ea    = (float*)carve((size_t)N_EDGES * HEADS * 4);
    float* XL    = (float*)carve((size_t)N_NODES * DMODEL * 4);
    float* AGG   = (float*)carve((size_t)N_NODES * DMODEL * 4);
    u16*   hbf   = (u16*)  carve((size_t)N_NODES * DMODEL * 2);
    float* asrc  = (float*)carve((size_t)N_NODES * HEADS * 4);
    float* adst  = (float*)carve((size_t)N_NODES * HEADS * 4);
    float* amax  = (float*)carve((size_t)N_NODES * HEADS * 4);
    float* denom = (float*)carve((size_t)N_NODES * HEADS * 4);
    float* alpha = (float*)carve((size_t)N_EDGES * HEADS * 4);
    float* logit = (float*)carve((size_t)N_NODES * NCLS_P * 4);
    (void)ws_size; (void)in_sizes; (void)n_in; (void)out_size;

    const int TPB = 256;
    dim3 gEdgeWave((N_EDGES * 32 + TPB - 1) / TPB);
    dim3 gEdge((N_EDGES + TPB - 1) / TPB);
    dim3 gNodeWave((N_NODES * 32 + TPB - 1) / TPB);
    dim3 gNode((N_NODES + TPB - 1) / TPB);
    dim3 gemmGrid((N_NODES + TM - 1) / TM, DMODEL / TN);
    dim3 gemmGridC((N_NODES + TM - 1) / TM, 1);

    // ---- setup --------------------------------------------------------------
    k_f32_to_bf16<<<2048, TPB, 0, stream>>>(x, xbf, N_NODES * F_IN);
    k_transpose_bf16<<<128, TPB, 0, stream>>>(W1, W1t, F_IN, DMODEL, DMODEL);
    k_transpose_bf16<<<256, TPB, 0, stream>>>(W2, W2t, DMODEL, DMODEL, DMODEL);
    k_transpose_bf16<<<16,  TPB, 0, stream>>>(Wc, Wct, DMODEL, NCLS, NCLS_P);
    k_fold_M<<<1, 64, 0, stream>>>(We1, ae1, We2, ae2, Mf);
    k_edge_encode<<<gEdgeWave, TPB, 0, stream>>>(x, src, dst, Wee1, bee1, Wee2, bee2, ea, N_EDGES);

    // ---- layer 1 ------------------------------------------------------------
    k_wmma_gemm<<<gemmGrid, TPB, 0, stream>>>(xbf, W1t, XL, N_NODES, DMODEL, F_IN);
    k_node_alpha<<<gNodeWave, TPB, 0, stream>>>(XL, as1, ad1, asrc, adst, N_NODES);
    k_init<<<512, TPB, 0, stream>>>(amax, -3.0e38f, N_NODES * HEADS);
    k_init<<<512, TPB, 0, stream>>>(denom, 0.f, N_NODES * HEADS);
    k_init<<<2048, TPB, 0, stream>>>(AGG, 0.f, N_NODES * DMODEL);
    k_edge_alpha<<<gEdge, TPB, 0, stream>>>(src, dst, asrc, adst, ea, Mf, alpha, amax, N_EDGES);
    k_edge_exp<<<gEdge, TPB, 0, stream>>>(dst, alpha, amax, denom, N_EDGES);
    k_edge_agg<<<gEdgeWave, TPB, 0, stream>>>(src, dst, alpha, denom, XL, AGG, N_EDGES);
    k_bias_elu_bf16<<<2048, TPB, 0, stream>>>(AGG, b1, hbf, N_NODES * DMODEL);

    // ---- layer 2 ------------------------------------------------------------
    k_wmma_gemm<<<gemmGrid, TPB, 0, stream>>>(hbf, W2t, XL, N_NODES, DMODEL, DMODEL);
    k_node_alpha<<<gNodeWave, TPB, 0, stream>>>(XL, as2, ad2, asrc, adst, N_NODES);
    k_init<<<512, TPB, 0, stream>>>(amax, -3.0e38f, N_NODES * HEADS);
    k_init<<<512, TPB, 0, stream>>>(denom, 0.f, N_NODES * HEADS);
    k_init<<<2048, TPB, 0, stream>>>(AGG, 0.f, N_NODES * DMODEL);
    k_edge_alpha<<<gEdge, TPB, 0, stream>>>(src, dst, asrc, adst, ea, Mf + 16, alpha, amax, N_EDGES);
    k_edge_exp<<<gEdge, TPB, 0, stream>>>(dst, alpha, amax, denom, N_EDGES);
    k_edge_agg<<<gEdgeWave, TPB, 0, stream>>>(src, dst, alpha, denom, XL, AGG, N_EDGES);
    k_bias_elu_bf16<<<2048, TPB, 0, stream>>>(AGG, b2, hbf, N_NODES * DMODEL);

    // ---- classifier ---------------------------------------------------------
    k_wmma_gemm<<<gemmGridC, TPB, 0, stream>>>(hbf, Wct, logit, N_NODES, NCLS_P, DMODEL);
    k_logsoftmax<<<gNode, TPB, 0, stream>>>(logit, bc, out, N_NODES);
}
